// MambaNet_61160334295629
// MI455X (gfx1250) — matrix operand order repaired and checked by reading, not compiled, not run on hardware
//
#include <hip/hip_runtime.h>
#include <hip/hip_bf16.h>

// ---------------------------------------------------------------------------
// MambaNet forward for MI455X (gfx1250, wave32).
// GEMMs: v_wmma_f32_16x16x32_bf16, register-blocked 32x64 macro-tiles,
// N/K template-constant so epilogue addressing folds to immediate offsets.
// Scan: per-timestep operand row staged via global_load_async_to_lds_b32
// (ASYNCcnt path) on the serial critical chain.
// ---------------------------------------------------------------------------

#define BSZ     64
#define SEQ     100
#define IN_DIM  20
#define OUT_DIM 3
#define DMODEL  256
#define NLAYERS 4
#define DSTATE  16
#define DCONV   4
#define EDIM    512            // EXPAND * D_MODEL
#define DTRANK  16
#define NROWS   (BSZ * SEQ)    // 6400 (multiple of 32)
#define DBLW    (DTRANK + 2 * DSTATE)   // 48

typedef __bf16 bf16_t;
typedef __attribute__((ext_vector_type(16))) __bf16 v16bf;
typedef __attribute__((ext_vector_type(8)))  __bf16 v8bf;
typedef __attribute__((ext_vector_type(8)))  float  v8f;

// A fragment (16x32 bf16): lanes 0-15 hold K={0..15} of row (lane&15),
// lanes 16-31 hold K={16..31} -> two contiguous 16B loads.
static __device__ __forceinline__ v16bf load_afrag(const bf16_t* __restrict__ row,
                                                   int k0, int half)
{
    v8bf a0 = *(const v8bf*)(row + k0 + 16 * half);
    v8bf a1 = *(const v8bf*)(row + k0 + 8 + 16 * half);
    v16bf a;
    #pragma unroll
    for (int j = 0; j < 8; ++j) { a[j] = a0[j]; a[8 + j] = a1[j]; }
    return a;
}

// B fragment (32x16 bf16): lane n holds 16 contiguous K of column n
// (= row n of W since W is stored N x K) -> one contiguous 32B load.
static __device__ __forceinline__ v16bf load_bfrag(const bf16_t* __restrict__ row,
                                                   int k0, int half)
{
    return *(const v16bf*)(row + k0 + 16 * half);
}

// ---------------------------------------------------------------------------
// Register-blocked GEMM: C[M,N] (f32) = A[M,K] (bf16) @ W[N,K]^T (bf16).
// One wave -> 32x64 macro-tile (2 A frags x 4 B frags -> 8 WMMAs / K-step).
// N, K compile-time so all strides are immediate offsets.
// ---------------------------------------------------------------------------
template<int N, int K, bool ACC>
__global__ void __launch_bounds__(256)
gemm_bf16_wmma_32x64(const bf16_t* __restrict__ A, const bf16_t* __restrict__ W,
                     float* __restrict__ C, int M)
{
    static_assert(N % 64 == 0 && K % 32 == 0, "tile divisibility");
    const int lane = threadIdx.x & 31;
    const int wid  = blockIdx.x * (blockDim.x >> 5) + (threadIdx.x >> 5);
    constexpr int nStrips = N >> 6;              // 64-wide N strips
    const int mStrips = M >> 5;                  // 32-tall M strips
    if (wid >= mStrips * nStrips) return;        // wave-uniform exit
    const int ms   = wid / nStrips;
    const int ns   = wid % nStrips;
    const int half = lane >> 4;
    const int lr   = lane & 15;

    const bf16_t* __restrict__ Arow[2];
    Arow[0] = A + (ms * 32 + lr) * K;
    Arow[1] = Arow[0] + 16 * K;
    const bf16_t* __restrict__ Wrow[4];
    #pragma unroll
    for (int j = 0; j < 4; ++j)
        Wrow[j] = W + (ns * 64 + j * 16 + lr) * K;

    v8f acc[2][4];
    #pragma unroll
    for (int i = 0; i < 2; ++i)
        #pragma unroll
        for (int j = 0; j < 4; ++j)
            acc[i][j] = (v8f){};

    if (ACC) {
        #pragma unroll
        for (int i = 0; i < 2; ++i)
            #pragma unroll
            for (int j = 0; j < 4; ++j) {
                const float* Cb = C + (ms * 32 + i * 16 + 8 * half) * N
                                    + ns * 64 + j * 16 + lr;
                #pragma unroll
                for (int r = 0; r < 8; ++r)
                    acc[i][j][r] = Cb[r * N];
            }
    }

    for (int k0 = 0; k0 < K; k0 += 32) {
        v16bf a[2], b[4];
        #pragma unroll
        for (int i = 0; i < 2; ++i) a[i] = load_afrag(Arow[i], k0, half);
        #pragma unroll
        for (int j = 0; j < 4; ++j) b[j] = load_bfrag(Wrow[j], k0, half);
        #pragma unroll
        for (int i = 0; i < 2; ++i)
            #pragma unroll
            for (int j = 0; j < 4; ++j)
                acc[i][j] = __builtin_amdgcn_wmma_f32_16x16x32_bf16(
                    false, a[i], false, b[j], (short)0, acc[i][j], false, false);
    }

    #pragma unroll
    for (int i = 0; i < 2; ++i)
        #pragma unroll
        for (int j = 0; j < 4; ++j) {
            float* Cb = C + (ms * 32 + i * 16 + 8 * half) * N
                          + ns * 64 + j * 16 + lr;
            #pragma unroll
            for (int r = 0; r < 8; ++r)
                Cb[r * N] = acc[i][j][r];
        }
}

// ---------------------------------------------------------------------------
// Simple GEMM (one 16x16 tile per wave) for narrow N (x_proj: N=48).
// ---------------------------------------------------------------------------
template<int N, int K>
__global__ void __launch_bounds__(256)
gemm_bf16_wmma_16x16(const bf16_t* __restrict__ A, const bf16_t* __restrict__ W,
                     float* __restrict__ C, int M)
{
    static_assert(N % 16 == 0 && K % 32 == 0, "tile divisibility");
    const int lane = threadIdx.x & 31;
    const int wid  = blockIdx.x * (blockDim.x >> 5) + (threadIdx.x >> 5);
    constexpr int ntiles = N >> 4;
    const int mtiles = M >> 4;
    if (wid >= mtiles * ntiles) return;          // wave-uniform exit
    const int mt   = wid / ntiles;
    const int nt   = wid % ntiles;
    const int half = lane >> 4;
    const int lr   = lane & 15;

    const bf16_t* __restrict__ Arow = A + (mt * 16 + lr) * K;
    const bf16_t* __restrict__ Wrow = W + (nt * 16 + lr) * K;

    v8f acc = {};
    for (int k0 = 0; k0 < K; k0 += 32) {
        v16bf a = load_afrag(Arow, k0, half);
        v16bf b = load_bfrag(Wrow, k0, half);
        acc = __builtin_amdgcn_wmma_f32_16x16x32_bf16(
                  false, a, false, b, (short)0, acc, false, false);
    }

    float* Cb = C + (mt * 16 + 8 * half) * N + nt * 16 + lr;
    #pragma unroll
    for (int r = 0; r < 8; ++r)
        Cb[r * N] = acc[r];
}

// f32 -> bf16 weight conversion (run once per launch, negligible cost)
__global__ void __launch_bounds__(256)
cvt_bf16_kernel(const float* __restrict__ src, bf16_t* __restrict__ dst, int n)
{
    int idx = blockIdx.x * 256 + threadIdx.x;
    if (idx < n) dst[idx] = (bf16_t)src[idx];
}

// h = relu(x @ coding_W^T + b) + pos_emb[l]
__global__ void __launch_bounds__(DMODEL)
embed_kernel(const float* __restrict__ x, const float* __restrict__ cw,
             const float* __restrict__ cb, const float* __restrict__ pos,
             float* __restrict__ h)
{
    const int row = blockIdx.x;            // b*SEQ + l
    const int d   = threadIdx.x;
    const int l   = row % SEQ;
    const float* xr = x + (size_t)row * IN_DIM;
    const float* wr = cw + (size_t)d * IN_DIM;
    float s = cb[d];
    #pragma unroll
    for (int k = 0; k < IN_DIM; ++k) s += xr[k] * wr[k];
    s = fmaxf(s, 0.f) + pos[(size_t)l * DMODEL + d];
    h[(size_t)row * DMODEL + d] = s;
}

// xn = rmsnorm(h) * w, emitted as bf16 for the in_proj WMMA GEMM
__global__ void __launch_bounds__(DMODEL)
rmsnorm_to_bf16(const float* __restrict__ h, const float* __restrict__ w,
                bf16_t* __restrict__ out)
{
    __shared__ float red[DMODEL];
    const int row = blockIdx.x;
    const int t   = threadIdx.x;
    float v = h[(size_t)row * DMODEL + t];
    red[t] = v * v;
    __syncthreads();
    for (int s = DMODEL >> 1; s > 0; s >>= 1) {
        if (t < s) red[t] += red[t + s];
        __syncthreads();
    }
    float scale = rsqrtf(red[0] * (1.0f / DMODEL) + 1e-5f);
    out[(size_t)row * DMODEL + t] = (bf16_t)(v * scale * w[t]);
}

// Depthwise causal conv (4 taps) + bias + SiLU over xz[..., :EDIM].
// Writes f32 (for the scan) and bf16 (for the x_proj WMMA GEMM).
__global__ void __launch_bounds__(256)
conv_silu_kernel(const float* __restrict__ xz, const float* __restrict__ cw,
                 const float* __restrict__ cb, float* __restrict__ xin,
                 bf16_t* __restrict__ xinbf)
{
    const int idx = blockIdx.x * 256 + threadIdx.x;
    if (idx >= NROWS * EDIM) return;
    const int e  = idx & (EDIM - 1);
    const int bl = idx >> 9;               // EDIM == 512
    const int l  = bl % SEQ;
    const float* w = cw + (size_t)e * DCONV;
    float s = cb[e];
    #pragma unroll
    for (int t = 0; t < DCONV; ++t) {
        int ls = l - (DCONV - 1) + t;
        if (ls >= 0)
            s += w[t] * xz[(size_t)(bl - (DCONV - 1) + t) * (2 * EDIM) + e];
    }
    float sg = 1.f / (1.f + __expf(-s));
    s = s * sg;
    xin[idx]   = s;
    xinbf[idx] = (bf16_t)s;
}

// Fused: delta = softplus(dbl[:, :16] @ dtW^T + dtb), selective scan over L,
// y = (sum_n h*C + x*Dp) * silu(z). One block per batch, one thread per e.
// Per-timestep dbl row (48 floats) staged global->LDS via the CDNA5 async
// copy path (ASYNCcnt), issued per-lane by e<48, waited before the barrier.
__global__ void __launch_bounds__(EDIM)
scan_kernel(const float* __restrict__ xin, const float* __restrict__ dbl,
            const float* __restrict__ dtW, const float* __restrict__ dtb,
            const float* __restrict__ Alog, const float* __restrict__ Dp,
            const float* __restrict__ xz, bf16_t* __restrict__ ybf)
{
    const int b = blockIdx.x;
    const int e = threadIdx.x;

    float A[DSTATE], wdt[DSTATE], hst[DSTATE];
    #pragma unroll
    for (int n = 0; n < DSTATE; ++n) {
        A[n]   = -__expf(Alog[(size_t)e * DSTATE + n]);
        wdt[n] = dtW[(size_t)e * DSTATE + n];
        hst[n] = 0.f;
    }
    const float bias = dtb[e];
    const float dp   = Dp[e];

    __shared__ float sd[DBLW];
    const unsigned lds_dst = (unsigned)(uintptr_t)&sd[e & (64 - 1)];

    for (int l = 0; l < SEQ; ++l) {
        const size_t bl = (size_t)b * SEQ + l;
        __syncthreads();
        if (e < DBLW) {
            // async global->LDS copy, 4 bytes per active lane (ASYNCcnt)
            unsigned long long ga = (unsigned long long)(dbl + bl * DBLW + e);
            asm volatile("global_load_async_to_lds_b32 %0, %1, off"
                         :: "v"(lds_dst), "v"(ga) : "memory");
        }
        asm volatile("s_wait_asynccnt 0x0" ::: "memory");
        __syncthreads();

        if (l + 1 < SEQ) {  // prefetch next timestep (serial chain cover)
            __builtin_prefetch(xin + (bl + 1) * EDIM + e, 0, 1);
        }

        float ds = bias;
        #pragma unroll
        for (int n = 0; n < DTRANK; ++n) ds += sd[n] * wdt[n];
        float delta = (ds > 20.f) ? ds : log1pf(__expf(ds));

        const float xv = xin[bl * EDIM + e];
        float y = 0.f;
        #pragma unroll
        for (int n = 0; n < DSTATE; ++n) {
            float dA = __expf(delta * A[n]);
            hst[n] = dA * hst[n] + (delta * sd[DTRANK + n]) * xv;
            y += hst[n] * sd[DTRANK + DSTATE + n];
        }
        y += xv * dp;

        const float z = xz[bl * (2 * EDIM) + EDIM + e];
        y *= z / (1.f + __expf(-z));
        ybf[bl * EDIM + e] = (bf16_t)y;
    }
}

// out = relu(h) @ fcend_W^T + b    (N=3: scalar dot, WMMA not worth it)
__global__ void __launch_bounds__(256)
head_kernel(const float* __restrict__ h, const float* __restrict__ W,
            const float* __restrict__ bias, float* __restrict__ out)
{
    const int idx = blockIdx.x * 256 + threadIdx.x;
    if (idx >= NROWS * OUT_DIM) return;
    const int o   = idx % OUT_DIM;
    const int row = idx / OUT_DIM;
    const float* hr = h + (size_t)row * DMODEL;
    const float* wr = W + (size_t)o * DMODEL;
    float s = bias[o];
    for (int k = 0; k < DMODEL; ++k) s += fmaxf(hr[k], 0.f) * wr[k];
    out[idx] = s;
}

// ---------------------------------------------------------------------------
extern "C" void kernel_launch(void* const* d_in, const int* in_sizes, int n_in,
                              void* d_out, int out_size, void* d_ws, size_t ws_size,
                              hipStream_t stream)
{
    const float* x     = (const float*)d_in[0];
    const float* codW  = (const float*)d_in[1];
    const float* codB  = (const float*)d_in[2];
    const float* pos   = (const float*)d_in[3];
    const float* inW   = (const float*)d_in[4];
    const float* convW = (const float*)d_in[5];
    const float* convB = (const float*)d_in[6];
    const float* xpW   = (const float*)d_in[7];
    const float* dtW   = (const float*)d_in[8];
    const float* dtB   = (const float*)d_in[9];
    const float* Alog  = (const float*)d_in[10];
    const float* Dp    = (const float*)d_in[11];
    const float* outW  = (const float*)d_in[12];
    const float* normW = (const float*)d_in[13];
    const float* fcW   = (const float*)d_in[14];
    const float* fcB   = (const float*)d_in[15];
    float* out = (float*)d_out;

    // workspace carve-up (256B aligned slices)
    char* p = (char*)d_ws;
    auto alloc = [&](size_t bytes) -> char* {
        char* r = p;
        p += (bytes + 255) & ~(size_t)255;
        return r;
    };
    float*  hbuf   = (float*) alloc((size_t)NROWS * DMODEL * 4);
    float*  xzbuf  = (float*) alloc((size_t)NROWS * 2 * EDIM * 4);
    float*  xinf   = (float*) alloc((size_t)NROWS * EDIM * 4);
    float*  dblf   = (float*) alloc((size_t)NROWS * DBLW * 4);
    bf16_t* xnbf   = (bf16_t*)alloc((size_t)NROWS * DMODEL * 2);
    bf16_t* xinbf  = (bf16_t*)alloc((size_t)NROWS * EDIM * 2);
    bf16_t* ybf    = (bf16_t*)alloc((size_t)NROWS * EDIM * 2);
    bf16_t* winbf  = (bf16_t*)alloc((size_t)NLAYERS * 2 * EDIM * DMODEL * 2);
    bf16_t* wxpbf  = (bf16_t*)alloc((size_t)NLAYERS * DBLW * EDIM * 2);
    bf16_t* woutbf = (bf16_t*)alloc((size_t)NLAYERS * DMODEL * EDIM * 2);

    // one-time (per launch) weight conversion to bf16
    const int n_in_w = NLAYERS * 2 * EDIM * DMODEL;
    cvt_bf16_kernel<<<(n_in_w + 255) / 256, 256, 0, stream>>>(inW, winbf, n_in_w);
    const int n_xp = NLAYERS * DBLW * EDIM;
    cvt_bf16_kernel<<<(n_xp + 255) / 256, 256, 0, stream>>>(xpW, wxpbf, n_xp);
    const int n_ow = NLAYERS * DMODEL * EDIM;
    cvt_bf16_kernel<<<(n_ow + 255) / 256, 256, 0, stream>>>(outW, woutbf, n_ow);

    embed_kernel<<<NROWS, DMODEL, 0, stream>>>(x, codW, codB, pos, hbuf);

    for (int i = 0; i < NLAYERS; ++i) {
        rmsnorm_to_bf16<<<NROWS, DMODEL, 0, stream>>>(
            hbuf, normW + (size_t)i * DMODEL, xnbf);

        {   // in_proj: xz[6400,1024] = xn @ Win^T      (WMMA 32x64)
            int waves = (NROWS / 32) * ((2 * EDIM) / 64);
            gemm_bf16_wmma_32x64<2 * EDIM, DMODEL, false>
                <<<(waves + 7) / 8, 256, 0, stream>>>(
                    xnbf, winbf + (size_t)i * 2 * EDIM * DMODEL, xzbuf, NROWS);
        }

        {   // depthwise causal conv + SiLU
            int total = NROWS * EDIM;
            conv_silu_kernel<<<(total + 255) / 256, 256, 0, stream>>>(
                xzbuf, convW + (size_t)i * EDIM * DCONV,
                convB + (size_t)i * EDIM, xinf, xinbf);
        }

        {   // x_proj: dbl[6400,48] = xin @ Wxp^T       (WMMA 16x16, N=48)
            int waves = (NROWS / 16) * (DBLW / 16);
            gemm_bf16_wmma_16x16<DBLW, EDIM>
                <<<(waves + 7) / 8, 256, 0, stream>>>(
                    xinbf, wxpbf + (size_t)i * DBLW * EDIM, dblf, NROWS);
        }

        // fused dt-proj + softplus + selective scan + SiLU gate
        scan_kernel<<<BSZ, EDIM, 0, stream>>>(
            xinf, dblf,
            dtW  + (size_t)i * EDIM * DTRANK,
            dtB  + (size_t)i * EDIM,
            Alog + (size_t)i * EDIM * DSTATE,
            Dp   + (size_t)i * EDIM,
            xzbuf, ybf);

        {   // out_proj with fused residual: h += y @ Wout^T   (WMMA 32x64, acc)
            int waves = (NROWS / 32) * (DMODEL / 64);
            gemm_bf16_wmma_32x64<DMODEL, EDIM, true>
                <<<(waves + 7) / 8, 256, 0, stream>>>(
                    ybf, woutbf + (size_t)i * DMODEL * EDIM, hbuf, NROWS);
        }
    }

    {
        int total = NROWS * OUT_DIM;
        head_kernel<<<(total + 255) / 256, 256, 0, stream>>>(hbuf, fcW, fcB, out);
    }
}
